// DeCoDiT_6751688589470
// MI455X (gfx1250) — compile-verified, matrix operand order checked by reading, not run on hardware
//
#include <hip/hip_runtime.h>
#include <cmath>

typedef __attribute__((ext_vector_type(16))) _Float16 v16h;
typedef __attribute__((ext_vector_type(8)))  _Float16 v8h;
typedef __attribute__((ext_vector_type(8)))  float    v8f;

#define GF_NN    1
#define GF_SILU  2
#define GF_ACCUM 4
#define GF_MUL   8

#define BM 128
#define BN 64
#define BK 32
#define LDT 40   // LDS pitch in halves: 80B rows -> 16B aligned, 20-bank stride

// ---------------------------------------------------------------------------
// Generic GEMM:  C = epilogue( alpha * A ( B or B^T ) + bias )
//   A: [M,K] row-major (lda)
//   NT (NN=false): B [N,K] row-major (weights, lin(x,W)=x@W.T);  NN: B [K,N]
//   fp32 in/out, f16 WMMA compute with fp32 accumulation.
//   128x64 block tile, 8 waves, 32x32 per wave (4 WMMA per K-step),
//   double-buffered LDS. Compile-time specialization on operand order and
//   16B-alignment so the steady-state stager is branch-free b128 traffic.
// ---------------------------------------------------------------------------
template<bool NN, bool ALA, bool ALB>
__global__ __launch_bounds__(256) void gemm_wmma(
    const float* __restrict__ A, int lda,
    const float* __restrict__ B, int ldb,
    float* __restrict__ C, int ldc,
    const float* __restrict__ bias,
    int M, int Nn, int K, float alpha, int flags)
{
  __shared__ _Float16 sA[2][BM * LDT];
  __shared__ _Float16 sB[2][BN * LDT];
  const int tid  = threadIdx.x;
  const int wid  = tid >> 5;        // 8 waves
  const int lane = tid & 31;
  const int half = lane >> 4;       // 0/1
  const int idx  = lane & 15;       // row (A) / col (B,D) within 16-tile
  const int wr   = wid >> 1;        // 0..3 : 32-row slab
  const int wc   = wid & 1;         // 0..1 : 32-col slab
  const int m0   = blockIdx.y * BM;
  const int n0   = blockIdx.x * BN;

  const int rowT = tid >> 2;        // 0..63
  const int kkT  = (tid & 3) << 3;  // 0,8,16,24
  const int krB  = tid >> 3;        // 0..31 (NN path)
  const int nnB  = (tid & 7) << 3;  // 0..56 (NN path)

  v8f acc[2][2] = {};

  // ---- hoisted per-thread row bases (clamped to row 0 when OOB) ----
  bool aOk[2]; const float* aB[2];
#pragma unroll
  for (int rr = 0; rr < 2; ++rr) {
    const int gm = m0 + rowT + rr * 64;
    aOk[rr] = gm < M;
    aB[rr]  = A + (aOk[rr] ? (size_t)gm * lda : 0);
  }
  const bool bOkNT = (n0 + rowT) < Nn;
  const float* bBNT = B + (bOkNT ? (size_t)(n0 + rowT) * ldb : 0);
  // NN: per-element column validity/clamp (independent of k)
  bool nOk[8]; int nCl[8];
#pragma unroll
  for (int j = 0; j < 8; ++j) {
    const int gn = n0 + nnB + j;
    nOk[j] = gn < Nn;
    nCl[j] = nOk[j] ? gn : 0;
  }

  auto pack = [&](const float* va, bool ok, _Float16* dst) {
    v8h fr;
#pragma unroll
    for (int j = 0; j < 8; ++j) fr[j] = (_Float16)(ok ? va[j] : 0.f);
    *(v8h*)dst = fr;                                   // ds_store_b128
  };

  auto stageA = [&](int k0, _Float16* dst) {
    const bool kFull = (k0 + BK) <= K;                 // scalar-uniform
#pragma unroll
    for (int rr = 0; rr < 2; ++rr) {
      const float* src = aB[rr] + (k0 + kkT);
      float va[8];
      if (kFull) {
        if constexpr (ALA) {
          const float4 u0 = *(const float4*)src;       // global_load_b128
          const float4 u1 = *(const float4*)(src + 4);
          va[0] = u0.x; va[1] = u0.y; va[2] = u0.z; va[3] = u0.w;
          va[4] = u1.x; va[5] = u1.y; va[6] = u1.z; va[7] = u1.w;
        } else {
#pragma unroll
          for (int j = 0; j < 8; ++j) va[j] = src[j];  // base + imm offsets
        }
      } else {
#pragma unroll
        for (int j = 0; j < 8; ++j) {
          const int gk = k0 + kkT + j;
          const float t = aB[rr][gk < K ? gk : 0];
          va[j] = (gk < K) ? t : 0.f;
        }
      }
      pack(va, aOk[rr], &dst[(rowT + rr * 64) * LDT + kkT]);
    }
  };

  auto stageB = [&](int k0, _Float16* dst) {
    if constexpr (!NN) {
      const bool kFull = (k0 + BK) <= K;
      const float* src = bBNT + (k0 + kkT);
      float vb[8];
      if (kFull) {
        if constexpr (ALB) {
          const float4 u0 = *(const float4*)src;
          const float4 u1 = *(const float4*)(src + 4);
          vb[0] = u0.x; vb[1] = u0.y; vb[2] = u0.z; vb[3] = u0.w;
          vb[4] = u1.x; vb[5] = u1.y; vb[6] = u1.z; vb[7] = u1.w;
        } else {
#pragma unroll
          for (int j = 0; j < 8; ++j) vb[j] = src[j];
        }
      } else {
#pragma unroll
        for (int j = 0; j < 8; ++j) {
          const int gk = k0 + kkT + j;
          const float t = bBNT[gk < K ? gk : 0];
          vb[j] = (gk < K) ? t : 0.f;
        }
      }
      pack(vb, bOkNT, &dst[rowT * LDT + kkT]);
    } else {
      const bool kOk = (k0 + krB) < K;                 // per-thread, one select
      const float* src = B + (kOk ? (size_t)(k0 + krB) * ldb : 0);
#pragma unroll
      for (int j = 0; j < 8; ++j) {
        const float t = src[nCl[j]];
        dst[(nnB + j) * LDT + krB] = (_Float16)((kOk && nOk[j]) ? t : 0.f);
      }
    }
  };

  // fragment per documented 16-bit 16x32 layout: VGPR p<4: K=half*8+2p ; p>=4: +16
  auto frag = [&](const _Float16* base, int row) {
    v16h f;
#pragma unroll
    for (int p = 0; p < 8; ++p) {
      const int kp = (p < 4) ? (half * 8 + 2 * p) : (half * 8 + 2 * p + 8);
      const _Float16* q = &base[row * LDT + kp];
      f[2 * p] = q[0]; f[2 * p + 1] = q[1];
    }
    return f;
  };

  stageA(0, sA[0]);
  stageB(0, sB[0]);
  __syncthreads();
  int cur = 0;
  for (int k0 = 0; k0 < K; k0 += BK) {
    const int kn = k0 + BK;
    if (kn < K) { stageA(kn, sA[cur ^ 1]); stageB(kn, sB[cur ^ 1]); }
    const v16h af0 = frag(sA[cur], wr * 32 + idx);
    const v16h af1 = frag(sA[cur], wr * 32 + 16 + idx);
    const v16h bf0 = frag(sB[cur], wc * 32 + idx);
    const v16h bf1 = frag(sB[cur], wc * 32 + 16 + idx);
    acc[0][0] = __builtin_amdgcn_wmma_f32_16x16x32_f16(false, af0, false, bf0,
                                                       (short)0, acc[0][0], false, false);
    acc[0][1] = __builtin_amdgcn_wmma_f32_16x16x32_f16(false, af0, false, bf1,
                                                       (short)0, acc[0][1], false, false);
    acc[1][0] = __builtin_amdgcn_wmma_f32_16x16x32_f16(false, af1, false, bf0,
                                                       (short)0, acc[1][0], false, false);
    acc[1][1] = __builtin_amdgcn_wmma_f32_16x16x32_f16(false, af1, false, bf1,
                                                       (short)0, acc[1][1], false, false);
    __syncthreads();
    cur ^= 1;
  }

  // ---- epilogue: D layout lane(idx)=N col, VGPR j -> M = half*8 + j
#pragma unroll
  for (int a = 0; a < 2; ++a) {
#pragma unroll
    for (int tb = 0; tb < 2; ++tb) {
      const int n = n0 + wc * 32 + tb * 16 + idx;
      if (n >= Nn) continue;
      const float bv = bias ? bias[n] : 0.f;
#pragma unroll
      for (int j = 0; j < 8; ++j) {
        const int m = m0 + wr * 32 + a * 16 + half * 8 + j;
        if (m >= M) continue;
        float v = alpha * acc[a][tb][j] + bv;
        if (flags & GF_SILU) v = v / (1.f + __expf(-v));
        const size_t o = (size_t)m * ldc + n;
        if (flags & GF_MUL)   v *= C[o];
        if (flags & GF_ACCUM) v += C[o];
        C[o] = v;
      }
    }
  }
}

// ---------------------------------------------------------------------------
// Softmax over rows (scale already applied in QK GEMM via alpha)
// ---------------------------------------------------------------------------
__global__ void k_softmax(float* data, int L, int ld)
{
  float* p = data + (size_t)blockIdx.x * ld;
  __shared__ float red[256];
  const int t = threadIdx.x;
  float mx = -1e30f;
  for (int i = t; i < L; i += 256) mx = fmaxf(mx, p[i]);
  red[t] = mx; __syncthreads();
  for (int s = 128; s > 0; s >>= 1) { if (t < s) red[t] = fmaxf(red[t], red[t + s]); __syncthreads(); }
  mx = red[0]; __syncthreads();
  float sum = 0.f;
  for (int i = t; i < L; i += 256) { float e = __expf(p[i] - mx); p[i] = e; sum += e; }
  red[t] = sum; __syncthreads();
  for (int s = 128; s > 0; s >>= 1) { if (t < s) red[t] += red[t + s]; __syncthreads(); }
  const float inv = 1.f / red[0];
  for (int i = t; i < L; i += 256) p[i] *= inv;
}

// out = rms(x,w) * (1+sc) + sh   (sc/sh from adaLN mod buffer, per row-group)
__global__ void k_rmsmod(const float* __restrict__ x, const float* __restrict__ w,
                         const float* __restrict__ modb, int modStride,
                         int shOff, int scOff, int rpg,
                         float* __restrict__ out, int rows, int D)
{
  const int r = blockIdx.x; if (r >= rows) return;
  const float* xr = x + (size_t)r * D;
  __shared__ float red[256];
  const int t = threadIdx.x;
  float ss = 0.f;
  for (int i = t; i < D; i += 256) { float v = xr[i]; ss += v * v; }
  red[t] = ss; __syncthreads();
  for (int s = 128; s > 0; s >>= 1) { if (t < s) red[t] += red[t + s]; __syncthreads(); }
  const float rs = rsqrtf(red[0] / (float)D + 1e-6f);
  const float* mr = modb ? modb + (size_t)(r / rpg) * modStride : nullptr;
  float* orow = out + (size_t)r * D;
  for (int i = t; i < D; i += 256) {
    float h = xr[i] * rs * w[i];
    if (mr) h = h * (1.f + mr[scOff + i]) + mr[shOff + i];
    orow[i] = h;
  }
}

// layernorm (optional affine) + optional adaLN mod ; D <= 64, blockDim = 64
__global__ void k_lnmod(const float* __restrict__ x, const float* __restrict__ w,
                        const float* __restrict__ b,
                        const float* __restrict__ modb, int modStride,
                        int shOff, int scOff, int rpg,
                        float* __restrict__ out, int rows, int D)
{
  const int r = blockIdx.x; if (r >= rows) return;
  const int t = threadIdx.x;
  __shared__ float red[64];
  const float* xr = x + (size_t)r * D;
  float v = (t < D) ? xr[t] : 0.f;
  red[t] = v; __syncthreads();
  for (int s = 32; s > 0; s >>= 1) { if (t < s) red[t] += red[t + s]; __syncthreads(); }
  const float mean = red[0] / (float)D; __syncthreads();
  const float d0 = (t < D) ? (v - mean) : 0.f;
  red[t] = d0 * d0; __syncthreads();
  for (int s = 32; s > 0; s >>= 1) { if (t < s) red[t] += red[t + s]; __syncthreads(); }
  const float var = red[0] / (float)D;
  float h = d0 * rsqrtf(var + 1e-6f);
  if (w) h = h * w[t] + b[t];
  if (modb) {
    const float* mr = modb + (size_t)(r / rpg) * modStride;
    h = h * (1.f + mr[scOff + t]) + mr[shOff + t];
  }
  if (t < D) out[(size_t)r * D + t] = h;
}

// per-head RMS norm over HD elems, in place. grid=(rows, nh), blockDim=128
__global__ void k_headrms(float* __restrict__ buf, const float* __restrict__ w,
                          int hd, int rowStride, int colOff)
{
  float* p = buf + (size_t)blockIdx.x * rowStride + colOff + blockIdx.y * hd;
  __shared__ float red[128];
  const int t = threadIdx.x;
  const float v = (t < hd) ? p[t] : 0.f;
  red[t] = v * v; __syncthreads();
  for (int s = 64; s > 0; s >>= 1) { if (t < s) red[t] += red[t + s]; __syncthreads(); }
  const float rs = rsqrtf(red[0] / (float)hd + 1e-6f);
  if (t < hd) p[t] = v * rs * w[t];
}

// out[r,d] += g[(r/rpg)*gs + go + d] * x[r,d]
__global__ void k_gated_add(float* __restrict__ out, const float* __restrict__ x,
                            const float* __restrict__ g, int rpg, int gs, int go,
                            int rows, int D)
{
  const int i = blockIdx.x * blockDim.x + threadIdx.x;
  if (i >= rows * D) return;
  const int r = i / D, d = i % D;
  out[i] += g[(size_t)(r / rpg) * gs + go + d] * x[i];
}

__global__ void k_add(const float* a, const float* b, float* o, int n)
{ int i = blockIdx.x * blockDim.x + threadIdx.x; if (i < n) o[i] = a[i] + b[i]; }

__global__ void k_silu(const float* a, float* o, int n)
{ int i = blockIdx.x * blockDim.x + threadIdx.x; if (i < n) { float v = a[i]; o[i] = v / (1.f + __expf(-v)); } }

__global__ void k_addpos(float* buf, const float* pos, int rows, int D, int period)
{
  const int i = blockIdx.x * blockDim.x + threadIdx.x;
  if (i >= rows * D) return;
  const int r = i / D, d = i % D;
  buf[i] += pos[(size_t)(r % period) * D + d];
}

// xp[b, n, c*4+p1*2+p2] = x[b, c, hp1*2+p1, hp2*2+p2], n = hp1*32+hp2
__global__ void k_patchify(const float* __restrict__ x, float* __restrict__ xp)
{
  const int i = blockIdx.x * blockDim.x + threadIdx.x;
  if (i >= 2 * 1024 * 12) return;
  const int b = i / (1024 * 12);
  const int n = (i / 12) % 1024;
  const int e = i % 12;
  const int c = e >> 2, p1 = (e >> 1) & 1, p2 = e & 1;
  const int hp1 = n >> 5, hp2 = n & 31;
  xp[i] = x[(((size_t)(b * 3 + c) * 64 + hp1 * 2 + p1)) * 64 + hp2 * 2 + p2];
}

// timestep sinusoidal embedding te[b, 0..255]
__global__ void k_timestep(const int* __restrict__ t, float* __restrict__ te)
{
  const int b = blockIdx.x, i = threadIdx.x;
  const int j = i & 127;
  const float f = __expf(-logf(10000.f) * (float)j / 128.f);
  const float a = (float)t[b] * f;
  te[b * 256 + i] = (i < 128) ? cosf(a) : sinf(a);
}

// pein[bn*4+p, 0..34] = { pix(3), fourier coords(32) }
__global__ void k_build_pein(const float* __restrict__ xp, float* __restrict__ pe)
{
  const int i = blockIdx.x * blockDim.x + threadIdx.x;
  if (i >= 8192 * 35) return;
  const int r = i / 35, col = i % 35;
  const int bn = r >> 2, p = r & 3;
  float v;
  if (col < 3) {
    v = xp[(size_t)bn * 12 + col * 4 + p];
  } else {
    const int j = col - 3;
    const int d = j >> 4, jj = j & 15, k = jj & 7;
    const int iy = p >> 1, ix = p & 1;
    const float coord = (d == 0) ? (ix ? 1.f : -1.f) : (iy ? 1.f : -1.f);
    const float a = coord * ldexpf(3.14159265358979f, k);
    v = (jj < 8) ? sinf(a) : cosf(a);
  }
  pe[i] = v;
}

// img[b,c,hp1*2+p1,hp2*2+p2] = out[((b*1024+hp1*32+hp2)*4+p1*2+p2)*3 + c]
__global__ void k_unpatchify(const float* __restrict__ ob, float* __restrict__ img)
{
  const int i = blockIdx.x * blockDim.x + threadIdx.x;
  if (i >= 24576) return;
  const int b = i / 12288;
  const int c = (i / 4096) % 3;
  const int row = (i / 64) % 64;
  const int colp = i % 64;
  const int hp1 = row >> 1, p1 = row & 1, hp2 = colp >> 1, p2 = colp & 1;
  img[i] = ob[(((size_t)(b * 1024 + hp1 * 32 + hp2) * 4 + p1 * 2 + p2)) * 3 + c];
}

// ---------------------------------------------------------------------------
// Host-side GEMM dispatch: pick compile-time specialization by operand order
// and 16B alignment of each operand's row stride + base pointer.
// ---------------------------------------------------------------------------
static void launch_gemm(const float* A, int lda, const float* B, int ldb,
                        float* C, int ldc, const float* bias,
                        int M, int Nn, int K, float alpha, int flags,
                        hipStream_t stream)
{
  dim3 grid((Nn + BN - 1) / BN, (M + BM - 1) / BM);
  const bool alA = ((lda & 3) == 0) && ((((uintptr_t)A) & 15) == 0);
  const bool alB = ((ldb & 3) == 0) && ((((uintptr_t)B) & 15) == 0);
  if (flags & GF_NN) {
    if (alA) gemm_wmma<true, true, false><<<grid, 256, 0, stream>>>(A, lda, B, ldb, C, ldc, bias, M, Nn, K, alpha, flags);
    else     gemm_wmma<true, false, false><<<grid, 256, 0, stream>>>(A, lda, B, ldb, C, ldc, bias, M, Nn, K, alpha, flags);
  } else {
    if (alA && alB)      gemm_wmma<false, true,  true ><<<grid, 256, 0, stream>>>(A, lda, B, ldb, C, ldc, bias, M, Nn, K, alpha, flags);
    else if (alA)        gemm_wmma<false, true,  false><<<grid, 256, 0, stream>>>(A, lda, B, ldb, C, ldc, bias, M, Nn, K, alpha, flags);
    else if (alB)        gemm_wmma<false, false, true ><<<grid, 256, 0, stream>>>(A, lda, B, ldb, C, ldc, bias, M, Nn, K, alpha, flags);
    else                 gemm_wmma<false, false, false><<<grid, 256, 0, stream>>>(A, lda, B, ldb, C, ldc, bias, M, Nn, K, alpha, flags);
  }
}

// ---------------------------------------------------------------------------
extern "C" void kernel_launch(void* const* d_in, const int* in_sizes, int n_in,
                              void* d_out, int out_size, void* d_ws, size_t ws_size,
                              hipStream_t stream)
{
  (void)in_sizes; (void)n_in; (void)out_size; (void)ws_size;
  const float* x        = (const float*)d_in[0];
  const int*   t_in     = (const int*)  d_in[1];
  const float* context  = (const float*)d_in[2];
  const float* yv       = (const float*)d_in[3];
  const float* s_w      = (const float*)d_in[4];
  const float* s_b      = (const float*)d_in[5];
  const float* pos_emb  = (const float*)d_in[6];
  const float* t_w1     = (const float*)d_in[7];
  const float* t_b1     = (const float*)d_in[8];
  const float* t_w2     = (const float*)d_in[9];
  const float* t_b2     = (const float*)d_in[10];
  const float* lbl_w    = (const float*)d_in[11];
  const float* lbl_b    = (const float*)d_in[12];
  const float* y_w      = (const float*)d_in[13];
  const float* y_b      = (const float*)d_in[14];
  const float* y_pos    = (const float*)d_in[15];
  const float* tx_rms1  = (const float*)d_in[16];
  const float* tx_inw   = (const float*)d_in[17];
  const float* tx_inb   = (const float*)d_in[18];
  const float* tx_outw  = (const float*)d_in[19];
  const float* tx_outb  = (const float*)d_in[20];
  const float* tx_rms2  = (const float*)d_in[21];
  const float* tx_w1    = (const float*)d_in[22];
  const float* tx_b1    = (const float*)d_in[23];
  const float* tx_w2    = (const float*)d_in[24];
  const float* tx_b2    = (const float*)d_in[25];
  const float* tx_w3    = (const float*)d_in[26];
  const float* tx_b3    = (const float*)d_in[27];
  const float* tx_aw    = (const float*)d_in[28];
  const float* tx_ab    = (const float*)d_in[29];
  const float* en_rms1  = (const float*)d_in[30];
  const float* en_qkvw  = (const float*)d_in[31];
  const float* en_qn    = (const float*)d_in[32];
  const float* en_kn    = (const float*)d_in[33];
  const float* en_pw    = (const float*)d_in[34];
  const float* en_pb    = (const float*)d_in[35];
  const float* en_kvyw  = (const float*)d_in[36];
  const float* en_rms2  = (const float*)d_in[37];
  const float* en_w1    = (const float*)d_in[38];
  const float* en_b1    = (const float*)d_in[39];
  const float* en_w2    = (const float*)d_in[40];
  const float* en_b2    = (const float*)d_in[41];
  const float* en_w3    = (const float*)d_in[42];
  const float* en_b3    = (const float*)d_in[43];
  const float* en_aw    = (const float*)d_in[44];
  const float* en_ab    = (const float*)d_in[45];
  const float* ce_w     = (const float*)d_in[46];
  const float* ce_b     = (const float*)d_in[47];
  const float* dc_cw    = (const float*)d_in[48];
  const float* dc_cb    = (const float*)d_in[49];
  const float* dc_iw    = (const float*)d_in[50];
  const float* dc_ib    = (const float*)d_in[51];
  const float* dr_lnw   = (const float*)d_in[52];
  const float* dr_lnb   = (const float*)d_in[53];
  const float* dr_w1    = (const float*)d_in[54];
  const float* dr_b1    = (const float*)d_in[55];
  const float* dr_w2    = (const float*)d_in[56];
  const float* dr_b2    = (const float*)d_in[57];
  const float* dr_aw    = (const float*)d_in[58];
  const float* dr_ab    = (const float*)d_in[59];
  const float* dc_fw    = (const float*)d_in[60];
  const float* dc_fb    = (const float*)d_in[61];

  const int HID = 1152, MLP = 4608, NH = 16, HD = 72, LTOK = 100, TKV = 1124;
  const int Mimg = 2048;          // B*N
  const int Mtxt = 200;           // B*LTOK
  const float rs72 = 1.f / sqrtf(72.f);

  // -------- workspace carve-out (floats) --------
  float* W = (float*)d_ws;
  size_t off = 0;
  auto alloc = [&](size_t n) { float* p = W + off; off += n; return p; };
  float* xp      = alloc(2048 * 12);
  float* sbuf    = alloc((size_t)Mimg * HID);
  float* ctx     = alloc((size_t)Mtxt * HID);
  float* te0     = alloc(2 * 256);
  float* teh     = alloc(2 * HID);
  float* te      = alloc(2 * HID);
  float* lbl     = alloc(2 * HID);
  float* cvec    = alloc(2 * HID);
  float* csilu   = alloc(2 * HID);
  float* modb    = alloc(2 * 6 * HID);
  float* hbuf    = alloc((size_t)Mimg * HID);
  float* qkv     = alloc((size_t)Mimg * 3 * HID);
  float* kvy     = alloc((size_t)Mtxt * 2 * HID);
  float* scores  = alloc((size_t)1024 * TKV);
  float* ao      = alloc((size_t)Mimg * HID);
  float* tmp     = alloc((size_t)Mimg * HID);
  float* ubuf    = alloc((size_t)Mimg * MLP);
  float* pein    = alloc((size_t)8192 * 35);
  float* pe      = alloc((size_t)8192 * 64);
  float* hdec    = alloc((size_t)8192 * 64);
  float* cond    = alloc((size_t)8192 * 64);
  float* condsl  = alloc((size_t)8192 * 64);
  float* modr    = alloc((size_t)8192 * 192);
  float* u2      = alloc((size_t)8192 * 64);
  float* hh2     = alloc((size_t)8192 * 64);
  float* lnout   = alloc((size_t)8192 * 64);
  float* outb    = alloc((size_t)8192 * 3);

  auto GEMM = [&](const float* A, int lda, const float* Bm, int ldb,
                  float* Cm, int ldc, const float* bias,
                  int M, int Nn, int K, float alpha, int flags) {
    launch_gemm(A, lda, Bm, ldb, Cm, ldc, bias, M, Nn, K, alpha, flags, stream);
  };
  auto EW = [](int n) { return dim3((n + 255) / 256); };

  // ---- patchify + patch embed + pos ----
  k_patchify<<<EW(2048 * 12), 256, 0, stream>>>(x, xp);
  GEMM(xp, 12, s_w, 12, sbuf, HID, s_b, Mimg, HID, 12, 1.f, 0);
  k_addpos<<<EW(Mimg * HID), 256, 0, stream>>>(sbuf, pos_emb, Mimg, HID, 1024);

  // ---- conditioning vector c = MLP(t_emb) + label embed ----
  k_timestep<<<2, 256, 0, stream>>>(t_in, te0);
  GEMM(te0, 256, t_w1, 256, teh, HID, t_b1, 2, HID, 256, 1.f, GF_SILU);
  GEMM(teh, HID, t_w2, HID, te, HID, t_b2, 2, HID, HID, 1.f, 0);
  GEMM(yv, 4, lbl_w, 4, lbl, HID, lbl_b, 2, HID, 4, 1.f, 0);
  k_add<<<EW(2 * HID), 256, 0, stream>>>(te, lbl, cvec, 2 * HID);
  k_silu<<<EW(2 * HID), 256, 0, stream>>>(cvec, csilu, 2 * HID);

  // ---- text context embed ----
  GEMM(context, 1024, y_w, 1024, ctx, HID, y_b, Mtxt, HID, 1024, 1.f, 0);
  k_addpos<<<EW(Mtxt * HID), 256, 0, stream>>>(ctx, y_pos, Mtxt, HID, LTOK);

  // ================= text refine blocks =================
  for (int i = 0; i < 2; ++i) {
    GEMM(csilu, HID, tx_aw + (size_t)i * 6 * HID * HID, HID, modb, 6 * HID,
         tx_ab + (size_t)i * 6 * HID, 2, 6 * HID, HID, 1.f, 0);
    k_rmsmod<<<Mtxt, 256, 0, stream>>>(ctx, tx_rms1 + i * HID, modb, 6 * HID,
                                       0, HID, LTOK, hbuf, Mtxt, HID);
    GEMM(hbuf, HID, tx_inw + (size_t)i * 3 * HID * HID, HID, qkv, 3 * HID,
         tx_inb + (size_t)i * 3 * HID, Mtxt, 3 * HID, HID, 1.f, 0);
    for (int b = 0; b < 2; ++b)
      for (int h = 0; h < NH; ++h) {
        const float* base = qkv + (size_t)b * LTOK * 3 * HID + h * HD;
        GEMM(base, 3 * HID, base + HID, 3 * HID, scores, LTOK, nullptr,
             LTOK, LTOK, HD, rs72, 0);
        k_softmax<<<LTOK, 256, 0, stream>>>(scores, LTOK, LTOK);
        GEMM(scores, LTOK, base + 2 * HID, 3 * HID,
             ao + (size_t)b * LTOK * HID + h * HD, HID, nullptr,
             LTOK, HD, LTOK, 1.f, GF_NN);
      }
    GEMM(ao, HID, tx_outw + (size_t)i * HID * HID, HID, tmp, HID,
         tx_outb + (size_t)i * HID, Mtxt, HID, HID, 1.f, 0);
    k_gated_add<<<EW(Mtxt * HID), 256, 0, stream>>>(ctx, tmp, modb, LTOK, 6 * HID, 2 * HID, Mtxt, HID);
    k_rmsmod<<<Mtxt, 256, 0, stream>>>(ctx, tx_rms2 + i * HID, modb, 6 * HID,
                                       3 * HID, 4 * HID, LTOK, hbuf, Mtxt, HID);
    GEMM(hbuf, HID, tx_w1 + (size_t)i * MLP * HID, HID, ubuf, MLP,
         tx_b1 + (size_t)i * MLP, Mtxt, MLP, HID, 1.f, GF_SILU);
    GEMM(hbuf, HID, tx_w2 + (size_t)i * MLP * HID, HID, ubuf, MLP,
         tx_b2 + (size_t)i * MLP, Mtxt, MLP, HID, 1.f, GF_MUL);
    GEMM(ubuf, MLP, tx_w3 + (size_t)i * HID * MLP, MLP, tmp, HID,
         tx_b3 + (size_t)i * HID, Mtxt, HID, MLP, 1.f, 0);
    k_gated_add<<<EW(Mtxt * HID), 256, 0, stream>>>(ctx, tmp, modb, LTOK, 6 * HID, 5 * HID, Mtxt, HID);
  }

  // ================= DiT encoder blocks =================
  for (int i = 0; i < 6; ++i) {
    GEMM(csilu, HID, en_aw + (size_t)i * 6 * HID * HID, HID, modb, 6 * HID,
         en_ab + (size_t)i * 6 * HID, 2, 6 * HID, HID, 1.f, 0);
    k_rmsmod<<<Mimg, 256, 0, stream>>>(sbuf, en_rms1 + i * HID, modb, 6 * HID,
                                       0, HID, 1024, hbuf, Mimg, HID);
    GEMM(hbuf, HID, en_qkvw + (size_t)i * 3 * HID * HID, HID, qkv, 3 * HID,
         nullptr, Mimg, 3 * HID, HID, 1.f, 0);
    GEMM(ctx, HID, en_kvyw + (size_t)i * 2 * HID * HID, HID, kvy, 2 * HID,
         nullptr, Mtxt, 2 * HID, HID, 1.f, 0);
    k_headrms<<<dim3(Mimg, NH), 128, 0, stream>>>(qkv, en_qn + i * HD, HD, 3 * HID, 0);
    k_headrms<<<dim3(Mimg, NH), 128, 0, stream>>>(qkv, en_kn + i * HD, HD, 3 * HID, HID);
    k_headrms<<<dim3(Mtxt, NH), 128, 0, stream>>>(kvy, en_kn + i * HD, HD, 2 * HID, 0);
    for (int b = 0; b < 2; ++b)
      for (int h = 0; h < NH; ++h) {
        const float* qb  = qkv + (size_t)b * 1024 * 3 * HID + h * HD;
        const float* kyb = kvy + (size_t)b * LTOK * 2 * HID + h * HD;
        GEMM(qb, 3 * HID, qb + HID, 3 * HID, scores, TKV, nullptr,
             1024, 1024, HD, rs72, 0);
        GEMM(qb, 3 * HID, kyb, 2 * HID, scores + 1024, TKV, nullptr,
             1024, LTOK, HD, rs72, 0);
        k_softmax<<<1024, 256, 0, stream>>>(scores, TKV, TKV);
        float* aob = ao + (size_t)b * 1024 * HID + h * HD;
        GEMM(scores, TKV, qb + 2 * HID, 3 * HID, aob, HID, nullptr,
             1024, HD, 1024, 1.f, GF_NN);
        GEMM(scores + 1024, TKV, kyb + HID, 2 * HID, aob, HID, nullptr,
             1024, HD, LTOK, 1.f, GF_NN | GF_ACCUM);
      }
    GEMM(ao, HID, en_pw + (size_t)i * HID * HID, HID, tmp, HID,
         en_pb + (size_t)i * HID, Mimg, HID, HID, 1.f, 0);
    k_gated_add<<<EW(Mimg * HID), 256, 0, stream>>>(sbuf, tmp, modb, 1024, 6 * HID, 2 * HID, Mimg, HID);
    k_rmsmod<<<Mimg, 256, 0, stream>>>(sbuf, en_rms2 + i * HID, modb, 6 * HID,
                                       3 * HID, 4 * HID, 1024, hbuf, Mimg, HID);
    GEMM(hbuf, HID, en_w1 + (size_t)i * MLP * HID, HID, ubuf, MLP,
         en_b1 + (size_t)i * MLP, Mimg, MLP, HID, 1.f, GF_SILU);
    GEMM(hbuf, HID, en_w2 + (size_t)i * MLP * HID, HID, ubuf, MLP,
         en_b2 + (size_t)i * MLP, Mimg, MLP, HID, 1.f, GF_MUL);
    GEMM(ubuf, MLP, en_w3 + (size_t)i * HID * MLP, MLP, tmp, HID,
         en_b3 + (size_t)i * HID, Mimg, HID, MLP, 1.f, 0);
    k_gated_add<<<EW(Mimg * HID), 256, 0, stream>>>(sbuf, tmp, modb, 1024, 6 * HID, 5 * HID, Mimg, HID);
  }

  // ================= pixel decoder =================
  k_build_pein<<<EW(8192 * 35), 256, 0, stream>>>(xp, pein);
  GEMM(pein, 35, ce_w, 35, pe, 64, ce_b, 8192, 64, 35, 1.f, 0);
  GEMM(pe, 64, dc_iw, 64, hdec, 64, dc_ib, 8192, 64, 64, 1.f, 0);
  GEMM(sbuf, HID, dc_cw, HID, cond, 256, dc_cb, Mimg, 256, HID, 1.f, 0); // [8192,64] view
  k_silu<<<EW(8192 * 64), 256, 0, stream>>>(cond, condsl, 8192 * 64);
  for (int i = 0; i < 2; ++i) {
    GEMM(condsl, 64, dr_aw + (size_t)i * 192 * 64, 64, modr, 192,
         dr_ab + (size_t)i * 192, 8192, 192, 64, 1.f, 0);
    k_lnmod<<<8192, 64, 0, stream>>>(hdec, dr_lnw + i * 64, dr_lnb + i * 64,
                                     modr, 192, 0, 64, 1, lnout, 8192, 64);
    GEMM(lnout, 64, dr_w1 + (size_t)i * 64 * 64, 64, u2, 64,
         dr_b1 + (size_t)i * 64, 8192, 64, 64, 1.f, GF_SILU);
    GEMM(u2, 64, dr_w2 + (size_t)i * 64 * 64, 64, hh2, 64,
         dr_b2 + (size_t)i * 64, 8192, 64, 64, 1.f, 0);
    k_gated_add<<<EW(8192 * 64), 256, 0, stream>>>(hdec, hh2, modr, 1, 192, 128, 8192, 64);
  }
  k_lnmod<<<8192, 64, 0, stream>>>(hdec, nullptr, nullptr, nullptr, 0, 0, 0, 1,
                                   lnout, 8192, 64);
  GEMM(lnout, 64, dc_fw, 64, outb, 3, dc_fb, 8192, 3, 64, 1.f, 0);
  k_unpatchify<<<EW(24576), 256, 0, stream>>>(outb, (float*)d_out);
}
